// TemporalAttentionLayer_88587995447770
// MI455X (gfx1250) — matrix operand order repaired and checked by reading, not compiled, not run on hardware
//
#include <hip/hip_runtime.h>

// ---------------- problem dims ----------------
#define T_DIM 256
#define B_DIM 8
#define N_J   20
#define D_DIM 512
#define H_DIM 8
#define HD    64
#define M_ROWS (T_DIM * B_DIM)      // 2048
#define LDX   (N_J * D_DIM)         // 10240

typedef __bf16 bf16;
typedef __attribute__((ext_vector_type(16))) __bf16 v16bf;
typedef __attribute__((ext_vector_type(8)))  __bf16 v8bf;
typedef __attribute__((ext_vector_type(8)))  float  v8f;

// ---------------- WMMA helpers ----------------
__device__ inline v8f wmma_bf16(v16bf a, v16bf b, v8f c) {
  // D = A(16x32 bf16) * B(32x16 bf16) + C(16x16 f32)
  return __builtin_amdgcn_wmma_f32_16x16x32_bf16(
      /*neg_a=*/false, a, /*neg_b=*/false, b,
      /*c_mod=*/(short)0, c, /*reuse_a=*/false, /*reuse_b=*/false);
}

__device__ inline v8f zero8() {
  v8f z = {0.f, 0.f, 0.f, 0.f, 0.f, 0.f, 0.f, 0.f};
  return z;
}

// A fragment: matrix stored row-major (M x K), 16 rows, K-chunk of 32.
// lane&15 = row M; lanes 0-15 take K {0..7,16..23}, lanes 16-31 take {8..15,24..31}.
__device__ inline v16bf load_frag_a(const bf16* __restrict__ base, int ld) {
  int lane = threadIdx.x & 31;
  const bf16* p = base + (lane & 15) * ld + ((lane >> 4) << 3);
  v8bf lo = *(const v8bf*)(p);
  v8bf hi = *(const v8bf*)(p + 16);
  v16bf r;
#pragma unroll
  for (int i = 0; i < 8; ++i) { r[i] = lo[i]; r[i + 8] = hi[i]; }
  return r;
}

// B fragment: memory holds N x K row-major (i.e. B^T), representing B[k][n]=mem[n][k].
// lane&15 = column N; lanes 0-15 hold K {0..7,16..23}, lanes 16-31 hold K {8..15,24..31}.
__device__ inline v16bf load_frag_b(const bf16* __restrict__ base, int ld) {
  int lane = threadIdx.x & 31;
  const bf16* p = base + (lane & 15) * ld + ((lane >> 4) << 3);
  v8bf lo = *(const v8bf*)(p);
  v8bf hi = *(const v8bf*)(p + 16);
  v16bf r;
#pragma unroll
  for (int i = 0; i < 8; ++i) { r[i] = lo[i]; r[i + 8] = hi[i]; }
  return r;
}

// ---------------- fp32 -> bf16 convert ----------------
__global__ void cvt_f32_bf16(const float* __restrict__ s, bf16* __restrict__ d, int n) {
  int i = blockIdx.x * blockDim.x + threadIdx.x;
  if (i < n) d[i] = (bf16)s[i];
}

// ---------------- QKV projection ----------------
// grid: (M/64, D/64, N_J*3), block: 32 (one wave -> 64x64 tile)
__global__ __launch_bounds__(32) void proj_qkv(
    const bf16* __restrict__ xb,                       // (2048, 10240)
    const bf16* __restrict__ wq, const bf16* __restrict__ wk, const bf16* __restrict__ wv,
    const float* __restrict__ bq, const float* __restrict__ bk, const float* __restrict__ bv,
    bf16* __restrict__ qb, bf16* __restrict__ kb, bf16* __restrict__ vtb) {
  int mt = blockIdx.x * 64;
  int et = blockIdx.y * 64;
  int n  = blockIdx.z / 3;
  int wh = blockIdx.z % 3;

  const bf16*  W    = (wh == 0 ? wq : (wh == 1 ? wk : wv)) + (size_t)n * D_DIM * D_DIM;
  const float* bias = (wh == 0 ? bq : (wh == 1 ? bk : bv)) + n * D_DIM;
  const bf16*  A    = xb + n * D_DIM;   // joint-n columns; row stride LDX

  v8f acc[4][4];
#pragma unroll
  for (int i = 0; i < 4; ++i)
#pragma unroll
    for (int j = 0; j < 4; ++j) acc[i][j] = zero8();

  for (int k0 = 0; k0 < D_DIM; k0 += 32) {
    v16bf af[4], bfg[4];
#pragma unroll
    for (int i = 0; i < 4; ++i) af[i]  = load_frag_a(A + (mt + 16 * i) * LDX + k0, LDX);
#pragma unroll
    for (int j = 0; j < 4; ++j) bfg[j] = load_frag_b(W + (et + 16 * j) * D_DIM + k0, D_DIM);
#pragma unroll
    for (int i = 0; i < 4; ++i)
#pragma unroll
      for (int j = 0; j < 4; ++j) acc[i][j] = wmma_bf16(af[i], bfg[j], acc[i][j]);
  }

  int lane  = threadIdx.x & 31;
  int colL  = lane & 15;
  int rbase = (lane >> 4) * 8;
#pragma unroll
  for (int j = 0; j < 4; ++j) {
    int e = et + 16 * j + colL;
    float bb = bias[e];
    int h = e >> 6, f = e & 63;
#pragma unroll
    for (int i = 0; i < 4; ++i) {
#pragma unroll
      for (int r = 0; r < 8; ++r) {
        int m = mt + 16 * i + rbase + r;
        int t = m >> 3, b = m & 7;               // m = t*B + b
        float val = acc[i][j][r] + bb;
        int head = (b * N_J + n) * H_DIM + h;    // [b][n][h]
        if (wh == 0)      qb [(head << 14) + (t << 6) + f] = (bf16)val; // [t][f]
        else if (wh == 1) kb [(head << 14) + (t << 6) + f] = (bf16)val; // [t][f]
        else              vtb[(head << 14) + (f << 8) + t] = (bf16)val; // [f][t] transposed
      }
    }
  }
}

// ---------------- attention ----------------
// grid: (T/64, B*N*H), block: 128 (4 waves, each wave 16 query rows)
__global__ __launch_bounds__(128) void attn_kernel(
    const bf16* __restrict__ qb, const bf16* __restrict__ kb,
    const bf16* __restrict__ vtb, bf16* __restrict__ ob) {
  __shared__ float sc[4][16][T_DIM + 4];
  __shared__ float rowm[4][16], rowinv[4][16];

  int wave = threadIdx.x >> 5;
  int lane = threadIdx.x & 31;
  int hidx = blockIdx.y;                       // (b*N + n)*H + h
  int qt0  = blockIdx.x * 64 + wave * 16;

  const bf16* Q  = qb  + (hidx << 14) + qt0 * HD;
  const bf16* K  = kb  + (hidx << 14);
  const bf16* Vt = vtb + (hidx << 14);

  const float scale = 0.125f;                  // 1/sqrt(64)
  int colL  = lane & 15;
  int rbase = (lane >> 4) * 8;

  v16bf q0 = load_frag_a(Q, HD);
  v16bf q1 = load_frag_a(Q + 32, HD);

  // pass 1: scores + additive float tril mask
  for (int st = 0; st < 16; ++st) {
    v16bf k0 = load_frag_b(K + st * 16 * HD, HD);
    v16bf k1 = load_frag_b(K + st * 16 * HD + 32, HD);
    v8f s = zero8();
    s = wmma_bf16(q0, k0, s);
    s = wmma_bf16(q1, k1, s);
    int sg = st * 16 + colL;
#pragma unroll
    for (int r = 0; r < 8; ++r) {
      int tl = rbase + r;
      int tg = qt0 + tl;
      sc[wave][tl][sg] = s[r] * scale + (sg < tg ? 1.0f : 0.0f);
    }
  }
  __syncthreads();

  // softmax stats: two lanes per row, 128 cols each, combine via xor-16 shuffle
  {
    int row = lane & 15;
    int j0  = (lane >> 4) * 128;
    float m = -1e30f;
    for (int j = j0; j < j0 + 128; ++j) m = fmaxf(m, sc[wave][row][j]);
    m = fmaxf(m, __shfl_xor(m, 16, 32));
    float sum = 0.f;
    for (int j = j0; j < j0 + 128; ++j) sum += __expf(sc[wave][row][j] - m);
    sum += __shfl_xor(sum, 16, 32);
    if (lane < 16) { rowm[wave][row] = m; rowinv[wave][row] = 1.0f / sum; }
  }
  __syncthreads();

  // pass 2: O(16x64) = P(16x256) * V(256x64)
  v8f o[4];
#pragma unroll
  for (int j = 0; j < 4; ++j) o[j] = zero8();

  int row = lane & 15;
  float m  = rowm[wave][row];
  float iv = rowinv[wave][row];
  int klo = (lane >> 4) * 8;
  for (int st = 0; st < 8; ++st) {
    v16bf pf;
#pragma unroll
    for (int i = 0; i < 8; ++i) {
      int s0 = st * 32 + klo + i;
      int s1 = s0 + 16;
      pf[i]     = (bf16)(__expf(sc[wave][row][s0] - m) * iv);
      pf[i + 8] = (bf16)(__expf(sc[wave][row][s1] - m) * iv);
    }
#pragma unroll
    for (int j = 0; j < 4; ++j) {
      v16bf vf = load_frag_b(Vt + (j * 16) * T_DIM + st * 32, T_DIM);
      o[j] = wmma_bf16(pf, vf, o[j]);
    }
  }

  // store O into per-joint (M=2048, 512) bf16 matrix for the out-projection
  int h = hidx & 7;
  int n = (hidx >> 3) % N_J;
  int b = hidx / (N_J * H_DIM);
#pragma unroll
  for (int j = 0; j < 4; ++j) {
    int f = j * 16 + colL;
    int e = h * HD + f;
#pragma unroll
    for (int r = 0; r < 8; ++r) {
      int tg = qt0 + rbase + r;
      int mrow = tg * B_DIM + b;
      ob[(size_t)n * M_ROWS * D_DIM + (size_t)mrow * D_DIM + e] = (bf16)o[j][r];
    }
  }
}

// ---------------- out-projection ----------------
// grid: (M/64, D/64, N_J), block: 32
__global__ __launch_bounds__(32) void proj_out(
    const bf16* __restrict__ ob, const bf16* __restrict__ wo,
    const float* __restrict__ bo, float* __restrict__ out) {
  int mt = blockIdx.x * 64;
  int et = blockIdx.y * 64;
  int n  = blockIdx.z;

  const bf16* A = ob + (size_t)n * M_ROWS * D_DIM;
  const bf16* W = wo + (size_t)n * D_DIM * D_DIM;

  v8f acc[4][4];
#pragma unroll
  for (int i = 0; i < 4; ++i)
#pragma unroll
    for (int j = 0; j < 4; ++j) acc[i][j] = zero8();

  for (int k0 = 0; k0 < D_DIM; k0 += 32) {
    v16bf af[4], bfg[4];
#pragma unroll
    for (int i = 0; i < 4; ++i) af[i]  = load_frag_a(A + (mt + 16 * i) * D_DIM + k0, D_DIM);
#pragma unroll
    for (int j = 0; j < 4; ++j) bfg[j] = load_frag_b(W + (et + 16 * j) * D_DIM + k0, D_DIM);
#pragma unroll
    for (int i = 0; i < 4; ++i)
#pragma unroll
      for (int j = 0; j < 4; ++j) acc[i][j] = wmma_bf16(af[i], bfg[j], acc[i][j]);
  }

  int lane  = threadIdx.x & 31;
  int colL  = lane & 15;
  int rbase = (lane >> 4) * 8;
#pragma unroll
  for (int j = 0; j < 4; ++j) {
    int e = et + 16 * j + colL;
    float bb = bo[n * D_DIM + e];
#pragma unroll
    for (int i = 0; i < 4; ++i) {
#pragma unroll
      for (int r = 0; r < 8; ++r) {
        int m = mt + 16 * i + rbase + r;       // m = t*B + b
        out[(size_t)m * LDX + n * D_DIM + e] = acc[i][j][r] + bb;
      }
    }
  }
}

// ---------------- host launcher ----------------
extern "C" void kernel_launch(void* const* d_in, const int* in_sizes, int n_in,
                              void* d_out, int out_size, void* d_ws, size_t ws_size,
                              hipStream_t stream) {
  const float* x_f  = (const float*)d_in[0];
  const float* Wq_f = (const float*)d_in[1];
  const float* bq   = (const float*)d_in[2];
  const float* Wk_f = (const float*)d_in[3];
  const float* bk   = (const float*)d_in[4];
  const float* Wv_f = (const float*)d_in[5];
  const float* bv   = (const float*)d_in[6];
  const float* Wo_f = (const float*)d_in[7];
  const float* bo   = (const float*)d_in[8];
  float* out = (float*)d_out;

  const size_t nX = (size_t)M_ROWS * LDX;           // 20,971,520
  const size_t nW = (size_t)N_J * D_DIM * D_DIM;    //  5,242,880

  char* ws = (char*)d_ws;
  size_t off = 0;
  bf16* xb  = (bf16*)(ws + off); off += nX * sizeof(bf16);
  bf16* wqb = (bf16*)(ws + off); off += nW * sizeof(bf16);
  bf16* wkb = (bf16*)(ws + off); off += nW * sizeof(bf16);
  bf16* wvb = (bf16*)(ws + off); off += nW * sizeof(bf16);
  bf16* wob = (bf16*)(ws + off); off += nW * sizeof(bf16);
  bf16* qb  = (bf16*)(ws + off); off += nX * sizeof(bf16);
  bf16* kb  = (bf16*)(ws + off); off += nX * sizeof(bf16);
  bf16* vtb = (bf16*)(ws + off); off += nX * sizeof(bf16);
  bf16* ob  = (bf16*)(ws + off); off += nX * sizeof(bf16);

  // 1) fp32 -> bf16 conversions
  cvt_f32_bf16<<<(int)((nX + 255) / 256), 256, 0, stream>>>(x_f,  xb,  (int)nX);
  cvt_f32_bf16<<<(int)((nW + 255) / 256), 256, 0, stream>>>(Wq_f, wqb, (int)nW);
  cvt_f32_bf16<<<(int)((nW + 255) / 256), 256, 0, stream>>>(Wk_f, wkb, (int)nW);
  cvt_f32_bf16<<<(int)((nW + 255) / 256), 256, 0, stream>>>(Wv_f, wvb, (int)nW);
  cvt_f32_bf16<<<(int)((nW + 255) / 256), 256, 0, stream>>>(Wo_f, wob, (int)nW);

  // 2) Q/K/V projections (V stored transposed per head)
  dim3 gProj(M_ROWS / 64, D_DIM / 64, N_J * 3);
  proj_qkv<<<gProj, 32, 0, stream>>>(xb, wqb, wkb, wvb, bq, bk, bv, qb, kb, vtb);

  // 3) attention per (b, n, h)
  dim3 gAttn(T_DIM / 64, B_DIM * N_J * H_DIM);
  attn_kernel<<<gAttn, 128, 0, stream>>>(qb, kb, vtb, ob);

  // 4) out-projection (f32 output + bias)
  dim3 gOut(M_ROWS / 64, D_DIM / 64, N_J);
  proj_out<<<gOut, 32, 0, stream>>>(ob, wob, bo, out);
}
